// IntDense_57286273794902
// MI455X (gfx1250) — matrix operand order repaired and checked by reading, not compiled, not run on hardware
//
#include <hip/hip_runtime.h>
#include <hip/hip_bf16.h>

typedef __attribute__((ext_vector_type(16))) _Float16 v16h;
typedef __attribute__((ext_vector_type(8)))  _Float16 v8h;
typedef __attribute__((ext_vector_type(4)))  _Float16 v4h;
typedef __attribute__((ext_vector_type(8)))  float    v8f;
typedef __attribute__((ext_vector_type(4)))  float    f32x4;
typedef __attribute__((ext_vector_type(4)))  int      i32x4;

namespace {
constexpr int kTokens   = 8192;
constexpr int kDIn      = 4096;
constexpr int kFeatures = 16384;

constexpr int BM = 128;          // block tile M
constexpr int BN = 128;          // block tile N
constexpr int BK = 32;           // K per WMMA step
constexpr int LDA = BK + 8;      // 40 halves (80 B) per A-tile row, pad vs bank conflicts
constexpr int LDB = BN + 8;      // 136 halves (272 B) per B-tile row
}

// Y = (A @ Wint) * (1/scale[n]) + bias[n]
// A: f32 [8192,4096]  Wint: int32 codes 0..7 [4096,16384] (exact in f16)
__global__ __launch_bounds__(256)
void IntDense_57286273794902_kernel(const float* __restrict__ A,
                                    const int*   __restrict__ W,
                                    const float* __restrict__ scale,
                                    const float* __restrict__ bias,
                                    float* __restrict__ Y)
{
    __shared__ _Float16 Alds[BM * LDA];
    __shared__ _Float16 Blds[BK * LDB];

    const int tid  = threadIdx.x;
    const int lane = tid & 31;
    const int wave = tid >> 5;       // 8 waves
    const int wr   = wave & 3;       // 4 wave-rows  -> 32 M each
    const int wc   = wave >> 2;      // 2 wave-cols  -> 64 N each
    const int l16  = lane & 15;
    const int half = lane >> 4;      // lane group 0/1

    const int m0 = blockIdx.y * BM;
    const int n0 = blockIdx.x * BN;

    v8f acc[2][4];
    const v8f vzero = {};
#pragma unroll
    for (int i = 0; i < 2; ++i)
#pragma unroll
        for (int j = 0; j < 4; ++j) acc[i][j] = vzero;

    for (int k0 = 0; k0 < kDIn; k0 += BK) {
        __syncthreads();   // LDS consumed by previous iteration's fragment loads

        // ---- stage A tile: 128x32 f32 -> f16, 16 B per thread per step ----
#pragma unroll
        for (int i = 0; i < 4; ++i) {
            const int idx = tid + 256 * i;
            const int r   = idx >> 3;          // 0..127
            const int c   = (idx & 7) * 4;     // 0..28
            f32x4 v = *(const f32x4*)(A + (size_t)(m0 + r) * kDIn + k0 + c);
            v4h h = { (_Float16)v.x, (_Float16)v.y, (_Float16)v.z, (_Float16)v.w };
            *(v4h*)(Alds + r * LDA + c) = h;   // ds_store_b64
        }
        // ---- stage W tile: 32x128 int32 -> f16 (codes 0..7 are exact) ----
#pragma unroll
        for (int i = 0; i < 4; ++i) {
            const int idx = tid + 256 * i;
            const int r   = idx >> 5;          // 0..31
            const int c   = (idx & 31) * 4;    // 0..124
            i32x4 q = *(const i32x4*)(W + (size_t)(k0 + r) * kFeatures + n0 + c);
            v4h h = { (_Float16)q.x, (_Float16)q.y, (_Float16)q.z, (_Float16)q.w };
            *(v4h*)(Blds + r * LDB + c) = h;   // ds_store_b64
        }

        // ---- prefetch next K tile while this one is consumed ----
        if (k0 + BK < kDIn) {
            __builtin_prefetch(A + (size_t)(m0 + (tid >> 3)) * kDIn
                                 + (k0 + BK) + (tid & 7) * 4, 0, 1);
            __builtin_prefetch(W + (size_t)(k0 + BK + (tid >> 5)) * kFeatures
                                 + n0 + (tid & 31) * 4, 0, 1);
        }

        __syncthreads();

        // ---- A fragments: 16-bit 16x32 layout ----
        // lane l (M = l%16): VGPR0..3 hold K = 8*half .. 8*half+7,
        //                    VGPR4..7 hold K = 16+8*half .. 16+8*half+7
        v16h afrag[2];
#pragma unroll
        for (int mf = 0; mf < 2; ++mf) {
            const _Float16* base = Alds + (wr * 32 + mf * 16 + l16) * LDA;
            v8h lo = *(const v8h*)(base + 8 * half);        // ds_load_b128
            v8h hi = *(const v8h*)(base + 16 + 8 * half);   // ds_load_b128
            afrag[mf] = __builtin_shufflevector(lo, hi,
                0, 1, 2, 3, 4, 5, 6, 7, 8, 9, 10, 11, 12, 13, 14, 15);
        }
        // ---- B fragments: 16-bit 32x16 layout ----
        // lane l holds column N = l16; elems 0..7 K = 8*half..+7, elems 8..15 K = 16+8*half..+7
        v16h bfrag[4];
#pragma unroll
        for (int nf = 0; nf < 4; ++nf) {
            const _Float16* base =
                Blds + wc * 64 + nf * 16 + l16;
#pragma unroll
            for (int s = 0; s < 16; ++s) {
                const int kk = (s < 8) ? (8 * half + s) : (16 + 8 * half + (s - 8));
                bfrag[nf][s] = base[kk * LDB];
            }
        }

        // ---- 8x v_wmma_f32_16x16x32_f16 per K step ----
#pragma unroll
        for (int mf = 0; mf < 2; ++mf)
#pragma unroll
            for (int nf = 0; nf < 4; ++nf)
                acc[mf][nf] = __builtin_amdgcn_wmma_f32_16x16x32_f16(
                    false, afrag[mf], false, bfrag[nf],
                    (short)0, acc[mf][nf], false, false);
    }

    // ---- epilogue: C/D layout has M = d + 8*half, N = l16 ----
#pragma unroll
    for (int nf = 0; nf < 4; ++nf) {
        const int n = n0 + wc * 64 + nf * 16 + l16;
        const float inv = 1.0f / scale[n];
        const float b   = bias[n];
#pragma unroll
        for (int mf = 0; mf < 2; ++mf) {
            const int mbase = m0 + wr * 32 + mf * 16 + 8 * half;
#pragma unroll
            for (int d = 0; d < 8; ++d) {
                Y[(size_t)(mbase + d) * kFeatures + n] =
                    acc[mf][nf][d] * inv + b;
            }
        }
    }
}

extern "C" void kernel_launch(void* const* d_in, const int* in_sizes, int n_in,
                              void* d_out, int out_size, void* d_ws, size_t ws_size,
                              hipStream_t stream) {
    (void)in_sizes; (void)n_in; (void)d_ws; (void)ws_size; (void)out_size;
    const float* A = (const float*)d_in[0];   // inputs  [8192,4096] f32
    const int*   W = (const int*)  d_in[1];   // kernel  [4096,16384] int32 codes
    const float* s = (const float*)d_in[2];   // scale   [1,16384] f32
    const float* b = (const float*)d_in[3];   // bias    [16384] f32
    float*       Y = (float*)d_out;           // out     [8192,16384] f32

    dim3 grid(kFeatures / BN, kTokens / BM);  // (128, 64)
    IntDense_57286273794902_kernel<<<grid, dim3(256), 0, stream>>>(A, W, s, b, Y);
}
